// RoPEMultiheadAttention_78237124264315
// MI455X (gfx1250) — compile-verified, matrix operand order checked by reading, not run on hardware
//
#include <hip/hip_runtime.h>
#include <hip/hip_bf16.h>

// ---------------------------------------------------------------------------
// RoPE Multi-head Attention for MI455X (gfx1250).
// B=4 L=2048 E=1024 H=16 D=64.
// All matmuls: v_wmma_f32_16x16x32_bf16. LDS staging: global_load_async_to_lds
// (ASYNCcnt double-buffered). Softmax reductions: DPP (quad_perm/row_ror).
// ---------------------------------------------------------------------------

typedef __attribute__((ext_vector_type(16))) __bf16 v16bf;
typedef __attribute__((ext_vector_type(8)))  float  v8f;

union Frag16 {            // one 16x32 bf16 WMMA operand per lane: 32 bytes
    v16bf v;
    uint4 u[2];
};

static __device__ __forceinline__
v8f wmma_bf16(const Frag16& a, const Frag16& b, v8f c) {
    return __builtin_amdgcn_wmma_f32_16x16x32_bf16(
        /*neg_a=*/false, a.v, /*neg_b=*/false, b.v,
        /*c_mod=*/(short)0, c, /*reuse_a=*/false, /*reuse_b=*/false);
}

// Async DMA of 16 bytes global -> LDS (per-lane addresses, ASYNCcnt tracked).
// Generic LDS pointers keep the LDS byte offset in their low 32 bits.
static __device__ __forceinline__
void async_b128(const __bf16* gsrc, const __bf16* ldst) {
    uint32_t l = (uint32_t)(uintptr_t)ldst;
    asm volatile("global_load_async_to_lds_b128 %0, %1, off"
                 :: "v"(l), "v"(gsrc) : "memory");
}
#define WAIT_ASYNC_LE(n) asm volatile("s_wait_asynccnt " #n ::: "memory")
#define WAIT_DS0()       asm volatile("s_wait_dscnt 0" ::: "memory")

// ---- DPP 16-lane row reductions (halves of a wave32 stay independent) ------
template <int CTRL>
static __device__ __forceinline__ float dpp_f(float v) {
#if __has_builtin(__builtin_amdgcn_update_dpp)
    return __int_as_float(
        __builtin_amdgcn_update_dpp(0, __float_as_int(v), CTRL, 0xF, 0xF, true));
#else
    return v;
#endif
}
static __device__ __forceinline__ float row16_max(float v) {
#if __has_builtin(__builtin_amdgcn_update_dpp)
    v = fmaxf(v, dpp_f<0xB1>(v));    // quad_perm [1,0,3,2]
    v = fmaxf(v, dpp_f<0x4E>(v));    // quad_perm [2,3,0,1]
    v = fmaxf(v, dpp_f<0x124>(v));   // row_ror:4
    v = fmaxf(v, dpp_f<0x128>(v));   // row_ror:8
#else
    for (int m = 1; m < 16; m <<= 1) v = fmaxf(v, __shfl_xor(v, m, 32));
#endif
    return v;
}
static __device__ __forceinline__ float row16_sum(float v) {
#if __has_builtin(__builtin_amdgcn_update_dpp)
    v += dpp_f<0xB1>(v);
    v += dpp_f<0x4E>(v);
    v += dpp_f<0x124>(v);
    v += dpp_f<0x128>(v);
#else
    for (int m = 1; m < 16; m <<= 1) v += __shfl_xor(v, m, 32);
#endif
    return v;
}

// ---------------------------------------------------------------------------
// fp32 -> bf16 convert
// ---------------------------------------------------------------------------
__global__ void f32_to_bf16_kernel(const float* __restrict__ in,
                                   __bf16* __restrict__ out, int n) {
    int i = blockIdx.x * blockDim.x + threadIdx.x;
    if (i < n) out[i] = (__bf16)in[i];
}

// ---------------------------------------------------------------------------
// GEMM: C[M,N] = A[M,K] @ B[N,K]^T + bias[N], bf16 in, fp32 accumulate.
// Block tile 128x128, 256 threads (8 waves); each wave: 16x128 strip
// (1 A-frag, 8 B-frags, 8 WMMAs per 32-wide K step).
// A/B tiles staged by async DMA into double-buffered LDS (rows padded to
// 40 bf16 so every fragment read is two aligned ds_load_b128).
// ---------------------------------------------------------------------------
template <bool STORE_F32>
__global__ __launch_bounds__(256)
void gemm_bf16_wmma(const __bf16* __restrict__ A,
                    const __bf16* __restrict__ Bw,
                    const float*  __restrict__ bias,
                    void* __restrict__ Cout,
                    int M, int N, int K) {
    constexpr int BM = 128, BN = 128, BK = 32, LDSS = 40;
    __shared__ __align__(16) __bf16 As[2][BM * LDSS];
    __shared__ __align__(16) __bf16 Bs[2][BN * LDSS];

    const int tid  = threadIdx.x;
    const int wave = tid >> 5;
    const int lane = tid & 31;
    const int col  = lane & 15;
    const int lkb  = (lane & 16) ? 8 : 0;   // per-lane K base inside 32-tile
    const int bm   = blockIdx.x * BM;
    const int bn   = blockIdx.y * BN;

    // 4 async b128 per thread per K-step (2 for A, 2 for B)
    auto stage = [&](int buf, int k0) {
#pragma unroll
        for (int i = 0; i < 2; ++i) {
            int idx = tid + i * 256;
            int r = idx >> 2, c = idx & 3;
            async_b128(A + (size_t)(bm + r) * K + k0 + c * 8,
                       &As[buf][r * LDSS + c * 8]);
        }
#pragma unroll
        for (int i = 0; i < 2; ++i) {
            int idx = tid + i * 256;
            int r = idx >> 2, c = idx & 3;
            async_b128(Bw + (size_t)(bn + r) * K + k0 + c * 8,
                       &Bs[buf][r * LDSS + c * 8]);
        }
    };

    v8f acc[8] = {};
    const int nk = K / BK;
    stage(0, 0);

    for (int i = 0; i < nk; ++i) {
        const int buf = i & 1;
        if (i + 1 < nk) {                 // keep next tile in flight
            stage(buf ^ 1, (i + 1) * BK);
            WAIT_ASYNC_LE(4);             // current tile's 4 DMAs done
        } else {
            WAIT_ASYNC_LE(0);
        }
        __syncthreads();

        Frag16 af;
        const int arow = wave * 16 + col;
        af.u[0] = *(const uint4*)(&As[buf][arow * LDSS + lkb]);
        af.u[1] = *(const uint4*)(&As[buf][arow * LDSS + lkb + 16]);
#pragma unroll
        for (int nt = 0; nt < 8; ++nt) {
            Frag16 bf;
            const int brow = nt * 16 + col;
            bf.u[0] = *(const uint4*)(&Bs[buf][brow * LDSS + lkb]);
            bf.u[1] = *(const uint4*)(&Bs[buf][brow * LDSS + lkb + 16]);
            acc[nt] = wmma_bf16(af, bf, acc[nt]);
        }
        __syncthreads();
    }

    // epilogue: C-layout row = (lane>=16?8:0)+r, col = lane&15
    const int rbase = bm + wave * 16 + ((lane & 16) ? 8 : 0);
#pragma unroll
    for (int nt = 0; nt < 8; ++nt) {
        const int n  = bn + nt * 16 + col;
        const float bv = bias[n];
#pragma unroll
        for (int r = 0; r < 8; ++r) {
            float v = acc[nt][r] + bv;
            if (STORE_F32)
                ((float*)Cout)[(size_t)(rbase + r) * N + n] = v;
            else
                ((__bf16*)Cout)[(size_t)(rbase + r) * N + n] = (__bf16)v;
        }
    }
}

// ---------------------------------------------------------------------------
// RoPE + head split. qkv: (B,L,3E) bf16. Outputs:
//   q,k : (B*H, L, D) bf16 (RoPE applied)
//   vT  : (B*H, D, L) bf16 (transposed so P@V B-operand is contiguous)
// ---------------------------------------------------------------------------
__global__ __launch_bounds__(256)
void rope_reshape_kernel(const __bf16* __restrict__ qkv,
                         const float*  __restrict__ freqs,
                         __bf16* __restrict__ q,
                         __bf16* __restrict__ k,
                         __bf16* __restrict__ vt) {
    const int idx = blockIdx.x * blockDim.x + threadIdx.x;
    const int d = idx & 63;
    const int h = (idx >> 6) & 15;
    const int l = (idx >> 10) & 2047;
    const int b = idx >> 21;

    const size_t base = ((size_t)(b * 2048 + l)) * 3072 + h * 64;
    float qv = (float)qkv[base + d];
    float kv = (float)qkv[base + 1024 + d];
    float vv = (float)qkv[base + 2048 + d];

    const int dp = (d < 32) ? d + 32 : d - 32;
    float qr = (float)qkv[base + dp];
    float kr = (float)qkv[base + 1024 + dp];
    if (d < 32) { qr = -qr; kr = -kr; }

    const float f = freqs[l * 64 + d];
    const float c = __cosf(f), s = __sinf(f);

    const int bh = b * 16 + h;
    const size_t o = ((size_t)bh * 2048 + l) * 64 + d;
    q[o] = (__bf16)(qv * c + qr * s);
    k[o] = (__bf16)(kv * c + kr * s);
    vt[((size_t)bh * 64 + d) * 2048 + l] = (__bf16)vv;
}

// ---------------------------------------------------------------------------
// Flash-style attention. Grid (BH=64, L/128=16), 256 threads (8 waves),
// each wave owns 16 query rows (128-query tile amortizes K/V staging 2x).
// 64-key blocks:
//   K/V staged by async DMA into double-buffered LDS (shared by all 8 waves),
//   S = Q K^T: 8 WMMA; online softmax with DPP row reductions;
//   P via per-wave LDS (C-layout -> A-layout); O += P V: 8 WMMA.
// ---------------------------------------------------------------------------
__global__ __launch_bounds__(256)
void attn_kernel(const __bf16* __restrict__ Q,
                 const __bf16* __restrict__ K,
                 const __bf16* __restrict__ VT,
                 __bf16* __restrict__ O) {
    constexpr int L = 2048, D = 64, KB = 64, ST = 72; // ST: padded row (bf16)
    __shared__ __align__(16) __bf16 Ks[2][KB * ST];   // rows = key, cols = d
    __shared__ __align__(16) __bf16 Vs[2][D * ST];    // rows = d, cols = key
    __shared__ __align__(16) __bf16 Ps[8][16 * ST];   // per-wave P scratch

    const int bh   = blockIdx.x;          // 0..63
    const int q0   = blockIdx.y * 128;    // query tile base
    const int wave = threadIdx.x >> 5;
    const int lane = threadIdx.x & 31;
    const int col  = lane & 15;
    const int lkb  = (lane & 16) ? 8 : 0;
    const int half = (lane >> 4) * 8;     // C-layout row base for this lane

    // 4 async b128 per thread per key block (2 K + 2 V)
    auto stage = [&](int buf, int kb0) {
#pragma unroll
        for (int it = 0; it < 2; ++it) {
            int idx = threadIdx.x + it * 256;
            int r = idx >> 3, c = idx & 7;
            async_b128(K  + ((size_t)bh * L + kb0 + r) * D + c * 8,
                       &Ks[buf][r * ST + c * 8]);
            async_b128(VT + ((size_t)bh * D + r) * L + kb0 + c * 8,
                       &Vs[buf][r * ST + c * 8]);
        }
    };

    // Q fragments for d0 = 0, 32 (A-layout, straight from global)
    Frag16 qf[2];
    const __bf16* qp = Q + ((size_t)bh * L + q0 + wave * 16 + col) * D;
#pragma unroll
    for (int dd = 0; dd < 2; ++dd) {
        qf[dd].u[0] = *(const uint4*)(qp + dd * 32 + lkb);
        qf[dd].u[1] = *(const uint4*)(qp + dd * 32 + lkb + 16);
    }

    float m_i[8], l_i[8];
    v8f oacc[4] = {};
#pragma unroll
    for (int r = 0; r < 8; ++r) { m_i[r] = -1e30f; l_i[r] = 0.0f; }

    const float scale = 0.125f;   // D^-0.5
    __bf16* ps = &Ps[wave][0];

    stage(0, 0);
    for (int i = 0; i < L / KB; ++i) {
        const int buf = i & 1;
        if (i + 1 < L / KB) {
            stage(buf ^ 1, (i + 1) * KB);
            WAIT_ASYNC_LE(4);
        } else {
            WAIT_ASYNC_LE(0);
        }
        __syncthreads();

        // ---- S = Q @ K^T over 64 keys: 4 n-tiles x 2 d-steps
        v8f s[4] = {};
#pragma unroll
        for (int nt = 0; nt < 4; ++nt) {
            const __bf16* kp = &Ks[buf][(nt * 16 + col) * ST];
#pragma unroll
            for (int dd = 0; dd < 2; ++dd) {
                Frag16 kf;
                kf.u[0] = *(const uint4*)(kp + dd * 32 + lkb);
                kf.u[1] = *(const uint4*)(kp + dd * 32 + lkb + 16);
                s[nt] = wmma_bf16(qf[dd], kf, s[nt]);
            }
        }

        // ---- online softmax (DPP row reductions, rows live in 16-lane groups)
        float alpha[8];
#pragma unroll
        for (int r = 0; r < 8; ++r) {
            float v = fmaxf(fmaxf(s[0][r], s[1][r]),
                            fmaxf(s[2][r], s[3][r])) * scale;
            v = row16_max(v);
            const float mn = fmaxf(m_i[r], v);
            alpha[r] = __expf(m_i[r] - mn);
            m_i[r] = mn;
        }

        float rs[8] = {0, 0, 0, 0, 0, 0, 0, 0};
#pragma unroll
        for (int nt = 0; nt < 4; ++nt)
#pragma unroll
            for (int r = 0; r < 8; ++r) {
                const float p = __expf(s[nt][r] * scale - m_i[r]);
                rs[r] += p;
                ps[(half + r) * ST + nt * 16 + col] = (__bf16)p;
            }
#pragma unroll
        for (int r = 0; r < 8; ++r) {
            l_i[r] = l_i[r] * alpha[r] + row16_sum(rs[r]);
#pragma unroll
            for (int t = 0; t < 4; ++t)
                oacc[t][r] *= alpha[r];
        }

        // wave-private P region: stores must land before A-layout reads
        WAIT_DS0();
        Frag16 pf[2];
#pragma unroll
        for (int kk = 0; kk < 2; ++kk) {
            pf[kk].u[0] = *(const uint4*)(&ps[col * ST + kk * 32 + lkb]);
            pf[kk].u[1] = *(const uint4*)(&ps[col * ST + kk * 32 + lkb + 16]);
        }

        // ---- O += P @ V : 4 d-tiles x 2 key-steps
#pragma unroll
        for (int t = 0; t < 4; ++t) {
            const __bf16* vp = &Vs[buf][(t * 16 + col) * ST];
#pragma unroll
            for (int kk = 0; kk < 2; ++kk) {
                Frag16 vf;
                vf.u[0] = *(const uint4*)(vp + kk * 32 + lkb);
                vf.u[1] = *(const uint4*)(vp + kk * 32 + lkb + 16);
                oacc[t] = wmma_bf16(pf[kk], vf, oacc[t]);
            }
        }
        __syncthreads();
    }

    // ---- normalize + store (B,L,E) bf16
    const int b = bh >> 4, h = bh & 15;
#pragma unroll
    for (int t = 0; t < 4; ++t)
#pragma unroll
        for (int r = 0; r < 8; ++r) {
            const int row = q0 + wave * 16 + half + r;
            const float v = oacc[t][r] / l_i[r];
            O[((size_t)(b * L + row)) * 1024 + h * 64 + t * 16 + col] = (__bf16)v;
        }
}

// ---------------------------------------------------------------------------
// Host-side orchestration
// ---------------------------------------------------------------------------
extern "C" void kernel_launch(void* const* d_in, const int* in_sizes, int n_in,
                              void* d_out, int out_size, void* d_ws, size_t ws_size,
                              hipStream_t stream) {
    (void)in_sizes; (void)n_in; (void)out_size; (void)ws_size;
    const float* x     = (const float*)d_in[0];
    const float* freqs = (const float*)d_in[1];
    const float* w_qkv = (const float*)d_in[2];
    const float* b_qkv = (const float*)d_in[3];
    const float* w_out = (const float*)d_in[4];
    const float* b_out = (const float*)d_in[5];

    const int Bn = 4, L = 2048, E = 1024;
    const size_t ML = (size_t)Bn * L;                 // 8192 rows

    char* ws = (char*)d_ws;
    size_t off = 0;
    auto carve = [&](size_t bytes) -> void* {
        void* p = ws + off;
        off += (bytes + 255) & ~(size_t)255;
        return p;
    };
    __bf16* xb    = (__bf16*)carve(ML * E * 2);           // x in bf16
    __bf16* wqkvb = (__bf16*)carve((size_t)3 * E * E * 2);
    __bf16* woutb = (__bf16*)carve((size_t)E * E * 2);
    __bf16* qkv   = (__bf16*)carve(ML * 3 * E * 2);       // (B,L,3E)
    __bf16* qb    = (__bf16*)carve(ML * E * 2);           // (BH,L,D)
    __bf16* kbuf  = (__bf16*)carve(ML * E * 2);           // (BH,L,D)
    __bf16* vtb   = (__bf16*)carve(ML * E * 2);           // (BH,D,L)
    __bf16* ao    = (__bf16*)carve(ML * E * 2);           // (B,L,E)

    // 1) fp32 -> bf16 converts
    {
        int n = (int)(ML * E);
        f32_to_bf16_kernel<<<(n + 255) / 256, 256, 0, stream>>>(x, xb, n);
        n = 3 * E * E;
        f32_to_bf16_kernel<<<(n + 255) / 256, 256, 0, stream>>>(w_qkv, wqkvb, n);
        n = E * E;
        f32_to_bf16_kernel<<<(n + 255) / 256, 256, 0, stream>>>(w_out, woutb, n);
    }

    // 2) QKV projection: (8192,1024) @ (3072,1024)^T + b_qkv -> bf16
    gemm_bf16_wmma<false><<<dim3(8192 / 128, 3072 / 128), 256, 0, stream>>>(
        xb, wqkvb, b_qkv, qkv, 8192, 3072, 1024);

    // 3) RoPE + head split (+ V transpose)
    rope_reshape_kernel<<<(int)(ML * E / 256), 256, 0, stream>>>(
        qkv, freqs, qb, kbuf, vtb);

    // 4) attention: 128-query tiles, 8 waves
    attn_kernel<<<dim3(64, 16), 256, 0, stream>>>(qb, kbuf, vtb, ao);

    // 5) output projection: (8192,1024) @ (1024,1024)^T + b_out -> fp32 d_out
    gemm_bf16_wmma<true><<<dim3(8192 / 128, 1024 / 128), 256, 0, stream>>>(
        ao, woutb, b_out, d_out, 8192, 1024, 1024);
}
